// FusionGeneGNN_28467043238119
// MI455X (gfx1250) — matrix operand order repaired and checked by reading, not compiled, not run on hardware
//
#include <hip/hip_runtime.h>
#include <hip/hip_bf16.h>

#define E_DIM 256
#define DH 64

typedef __attribute__((ext_vector_type(16))) __bf16 v16bf;
typedef __attribute__((ext_vector_type(8)))  __bf16 v8bf;
typedef __attribute__((ext_vector_type(8)))  float  v8f;

// fp32 -> bf16 round-to-nearest-even (elementwise passes only)
__device__ __forceinline__ __bf16 f2bf(float f) {
  unsigned u = __builtin_bit_cast(unsigned, f);
  u += 0x7FFFu + ((u >> 16) & 1u);
  unsigned short h = (unsigned short)(u >> 16);
  return __builtin_bit_cast(__bf16, h);
}

struct BfPair { v8bf lo, hi; };

// A 16x32 bf16 fragment (contiguous k order, same as B): lane m=lane&15, half=lane>>4.
// f[0..15]=A[m][kb+half*16+0..15] -> two b128 loads.
__device__ __forceinline__ v16bf load_a_bf(const __bf16* __restrict__ arow, int kb) {
  BfPair p{*(const v8bf*)(arow + kb), *(const v8bf*)(arow + kb + 8)};
  return __builtin_bit_cast(v16bf, p);
}

// B 32x16 bf16 fragment from pre-transposed (N x K) weights: lane n=lane&15,
// lanes 0-15 hold K=kb..kb+15, lanes 16-31 hold K=kb+16..kb+31 -> contiguous 32B.
__device__ __forceinline__ v16bf load_b_bf(const __bf16* __restrict__ brow, int kb) {
  return *(const v16bf*)(brow + kb);
}

// ---- bf16 WMMA GEMM: C(16 x 32 per wave) = act(A @ Bt^T + bias) ----
template <bool OUTBF, bool RELU>
__global__ void __launch_bounds__(128)
gemm_bf16(const __bf16* __restrict__ A, int lda,
          const __bf16* __restrict__ Bt, int ldb,
          const float* __restrict__ bias,
          void* __restrict__ Cv, int ldc,
          int tilesM, int K) {
  const int lane  = threadIdx.x & 31;
  const int wave  = threadIdx.x >> 5;
  const int tileM = blockIdx.y * 4 + wave;
  if (tileM >= tilesM) return;
  const int rowBase = tileM * 16;
  const int nBase   = blockIdx.x * 32;      // two 16-col tiles per wave
  const int half = lane >> 4, lm = lane & 15;

  const __bf16* arow = A  + (size_t)(rowBase + lm) * lda + half * 16;
  const __bf16* b0   = Bt + (size_t)(nBase + lm) * ldb + half * 16;
  const __bf16* b1   = b0 + (size_t)16 * ldb;

  v8f acc0 = {}, acc1 = {};
  for (int kb = 0; kb < K; kb += 32) {
    v16bf a  = load_a_bf(arow, kb);
    v16bf f0 = load_b_bf(b0, kb);
    v16bf f1 = load_b_bf(b1, kb);
    acc0 = __builtin_amdgcn_wmma_f32_16x16x32_bf16(false, a, false, f0, (short)0, acc0, false, false);
    acc1 = __builtin_amdgcn_wmma_f32_16x16x32_bf16(false, a, false, f1, (short)0, acc1, false, false);
  }

  const int col0 = nBase + lm, col1 = col0 + 16;
  const float bv0 = bias ? bias[col0] : 0.0f;
  const float bv1 = bias ? bias[col1] : 0.0f;
  const int r0 = rowBase + half * 8;
#pragma unroll
  for (int r = 0; r < 8; ++r) {
    float v0 = acc0[r] + bv0, v1 = acc1[r] + bv1;
    if (RELU) { v0 = v0 > 0.f ? v0 : 0.f; v1 = v1 > 0.f ? v1 : 0.f; }
    if (OUTBF) {
      __bf16* C = (__bf16*)Cv;
      C[(size_t)(r0 + r) * ldc + col0] = f2bf(v0);
      C[(size_t)(r0 + r) * ldc + col1] = f2bf(v1);
    } else {
      float* C = (float*)Cv;
      C[(size_t)(r0 + r) * ldc + col0] = v0;
      C[(size_t)(r0 + r) * ldc + col1] = v1;
    }
  }
}

// ---- Fused QKV + 4-token MHA + token-mean. One block = 4 nodes (16 p-rows). ----
// The block's shared 16x256 bf16 A-tile of p is staged to LDS once via
// global_load_async_to_lds_b128 (ASYNCcnt), then all 8 waves read fragments
// from LDS instead of redundantly re-fetching from global.
__global__ void __launch_bounds__(256)
qkv_attn(const __bf16* __restrict__ p,     // (4N, 256) bf16
         const __bf16* __restrict__ Wt,    // in_proj_w (768, 256) bf16 (N x K)
         const float*  __restrict__ bqkv,  // (768,)
         __bf16* __restrict__ obar) {      // (N, 256) bf16 token-mean of attn out
  __shared__ __align__(16) __bf16 sA[16 * E_DIM];  // staged p tile (8 KB)
  __shared__ float sT[3][16 * E_DIM];              // q,k,v tiles fp32 (48 KB)
  __shared__ float sS[4 * 64];                     // scores [node][h*16+i*4+j]

  const int lane = threadIdx.x & 31;
  const int wave = threadIdx.x >> 5;
  const int rowBase = blockIdx.x * 16;
  const int half = lane >> 4, lm = lane & 15;

  // Phase 0: async-stage the contiguous 8 KB p tile into LDS.
  {
    const __bf16* gsrc = p + (size_t)rowBase * E_DIM;   // 16 rows, contiguous
    unsigned ldsBase = (unsigned)(size_t)(&sA[0]);
#pragma unroll
    for (int it = 0; it < 2; ++it) {
      const int c = threadIdx.x + it * 256;             // 16-byte chunk id, 512 total
      unsigned lds = ldsBase + c * 16;
      const __bf16* g = gsrc + c * 8;
      asm volatile("global_load_async_to_lds_b128 %0, %1, off"
                   :: "v"(lds), "v"(g) : "memory");
    }
    asm volatile("s_wait_asynccnt 0" ::: "memory");
  }
  __syncthreads();

  // Phase 1: q/k/v = p @ Wm^T + b -> LDS. 48 col-tiles over 8 waves, 6 each.
  // A fragment (from LDS) hoisted per K-chunk across the wave's 6 tasks.
  const __bf16* arow = &sA[lm * E_DIM + half * 16];
  const __bf16* bptr[6];
#pragma unroll
  for (int t = 0; t < 6; ++t) {
    const int task = wave * 6 + t;         // 0..47 -> row (mat*256 + tn*16 + n) of Wt
    bptr[t] = Wt + (size_t)(task * 16 + lm) * E_DIM + half * 16;
  }
  v8f acc[6] = {};
#pragma unroll
  for (int kb = 0; kb < E_DIM; kb += 32) {
    v16bf a = load_a_bf(arow, kb);
#pragma unroll
    for (int t = 0; t < 6; ++t)
      acc[t] = __builtin_amdgcn_wmma_f32_16x16x32_bf16(false, a, false,
                 load_b_bf(bptr[t], kb), (short)0, acc[t], false, false);
  }
#pragma unroll
  for (int t = 0; t < 6; ++t) {
    const int task = wave * 6 + t;
    const int mat = task >> 4, tn = task & 15;
    const int col = tn * 16 + lm;
    const float bv = bqkv[mat * E_DIM + col];
    const int r0 = half * 8;
#pragma unroll
    for (int r = 0; r < 8; ++r)
      sT[mat][(r0 + r) * E_DIM + col] = acc[t][r] + bv;
  }
  __syncthreads();

  // Phase 2: attention, 64 threads per node.
  const int ln = threadIdx.x >> 6;   // node in block; p-row = ln*4 + token
  const int u  = threadIdx.x & 63;

  { // scores[h][i][j] = dot(q[i,h,:], k[j,h,:]) / 8
    const int h = u >> 4, i = (u >> 2) & 3, j = u & 3;
    const float* qr = &sT[0][(ln * 4 + i) * E_DIM + h * DH];
    const float* kr = &sT[1][(ln * 4 + j) * E_DIM + h * DH];
    float s = 0.f;
#pragma unroll
    for (int d = 0; d < DH; ++d) s += qr[d] * kr[d];
    sS[ln * 64 + u] = s * 0.125f;
  }
  __syncthreads();

  if (u < 16) { // softmax over j
    float* sc = &sS[ln * 64 + (u >> 2) * 16 + (u & 3) * 4];
    float m = fmaxf(fmaxf(sc[0], sc[1]), fmaxf(sc[2], sc[3]));
    float e0 = __expf(sc[0] - m), e1 = __expf(sc[1] - m);
    float e2 = __expf(sc[2] - m), e3 = __expf(sc[3] - m);
    float inv = 1.0f / (e0 + e1 + e2 + e3);
    sc[0] = e0 * inv; sc[1] = e1 * inv; sc[2] = e2 * inv; sc[3] = e3 * inv;
  }
  __syncthreads();

  { // obar[n][h*64+d] = 0.25 * sum_i sum_j attn * v
    const int node = blockIdx.x * 4 + ln;
#pragma unroll
    for (int c = 0; c < 4; ++c) {
      float acc2 = 0.f;
#pragma unroll
      for (int j = 0; j < 4; ++j) {
        const float* a = &sS[ln * 64 + c * 16];
        float colsum = a[j] + a[4 + j] + a[8 + j] + a[12 + j];
        acc2 += colsum * sT[2][(ln * 4 + j) * E_DIM + c * DH + u];
      }
      obar[(size_t)node * E_DIM + c * DH + u] = f2bf(0.25f * acc2);
    }
  }
}

// ---- conversion passes ----
__global__ void conv_x(const float* __restrict__ x, __bf16* __restrict__ xb) {
  const int n = blockIdx.x;
  for (int c = threadIdx.x; c < 1024; c += 256)
    xb[(size_t)n * 1024 + c] = f2bf(x[(size_t)n * 1025 + c]);
}
// W (K x N) fp32 -> Wt (N x K) bf16 (transpose-convert)
__global__ void conv_wt(const float* __restrict__ W, __bf16* __restrict__ Wt, int K, int N) {
  int idx = blockIdx.x * 256 + threadIdx.x;
  if (idx < K * N) {
    int k = idx / N, n = idx % N;
    Wt[(size_t)n * K + k] = f2bf(W[idx]);
  }
}
// W already (N x K): straight convert
__global__ void conv_w(const float* __restrict__ W, __bf16* __restrict__ Wb, int count) {
  int idx = blockIdx.x * 256 + threadIdx.x;
  if (idx < count) Wb[idx] = f2bf(W[idx]);
}

// ---- small elementwise / graph kernels ----
__global__ void gw_token(const float* __restrict__ x, const float* __restrict__ wgw,
                         const float* __restrict__ bgw, __bf16* __restrict__ p) {
  const int n = blockIdx.x, e = threadIdx.x;
  float g = x[(size_t)n * 1025 + 1024];
  float v = g * wgw[e] + bgw[e];
  p[((size_t)n * 4 + 3) * E_DIM + e] = f2bf(v > 0.f ? v : 0.f);
}

__global__ void deg_init(float* __restrict__ deg, int n) {
  int i = blockIdx.x * blockDim.x + threadIdx.x;
  if (i < n) deg[i] = 1.0f;                       // self loop
}
__global__ void deg_count(const int* __restrict__ ei, int E, float* __restrict__ deg) {
  int e = blockIdx.x * blockDim.x + threadIdx.x;
  if (e < E) atomicAdd(&deg[ei[E + e]], 1.0f);    // dst row
}
__global__ void dinv_k(const float* __restrict__ deg, float* __restrict__ dinv, int n) {
  int i = blockIdx.x * blockDim.x + threadIdx.x;
  if (i < n) dinv[i] = rsqrtf(deg[i]);            // deg >= 1 always
}

__global__ void self_init(const float* __restrict__ hw, const float* __restrict__ dinv,
                          float* __restrict__ out, int F) {
  const int n = blockIdx.x, d = threadIdx.x;
  float s = dinv[n];
  out[(size_t)n * F + d] = s * s * hw[(size_t)n * F + d];
}

__global__ void edge_scatter(const int* __restrict__ ei, int E,
                             const float* __restrict__ hw, const float* __restrict__ dinv,
                             float* __restrict__ out, int F) {
  const int e = blockIdx.x, d = threadIdx.x;
  const int s = ei[e], t = ei[E + e];
  const float nv = dinv[s] * dinv[t];
  atomicAdd(&out[(size_t)t * F + d], nv * hw[(size_t)s * F + d]);
}

// finalize: h += bias, optional relu; optional bf16 mirror for the next GEMM's A
template <bool RELU>
__global__ void bias_act(float* __restrict__ h, const float* __restrict__ b,
                         __bf16* __restrict__ hb, int F) {
  const int n = blockIdx.x, d = threadIdx.x;
  float v = h[(size_t)n * F + d] + b[d];
  if (RELU) v = v > 0.f ? v : 0.f;
  h[(size_t)n * F + d] = v;
  if (hb) hb[(size_t)n * F + d] = f2bf(v);
}

extern "C" void kernel_launch(void* const* d_in, const int* in_sizes, int n_in,
                              void* d_out, int out_size, void* d_ws, size_t ws_size,
                              hipStream_t stream) {
  const float* x          = (const float*)d_in[0];
  const int*   ei         = (const int*)d_in[1];
  const float* w_go       = (const float*)d_in[2];
  const float* b_go       = (const float*)d_in[3];
  const float* w_re       = (const float*)d_in[4];
  const float* b_re       = (const float*)d_in[5];
  const float* w_me       = (const float*)d_in[6];
  const float* b_me       = (const float*)d_in[7];
  const float* w_gw       = (const float*)d_in[8];
  const float* b_gw       = (const float*)d_in[9];
  const float* in_proj_w  = (const float*)d_in[10];
  const float* in_proj_b  = (const float*)d_in[11];
  const float* out_proj_w = (const float*)d_in[12];
  const float* out_proj_b = (const float*)d_in[13];
  const float* w1         = (const float*)d_in[14];
  const float* b1         = (const float*)d_in[15];
  const float* w2         = (const float*)d_in[16];
  const float* b2         = (const float*)d_in[17];

  const int N = in_sizes[0] / 1025;   // 50000 (multiple of 16)
  const int E = in_sizes[1] / 2;      // 1600000
  float* out = (float*)d_out;
  (void)n_in; (void)out_size; (void)ws_size;

  char* ws = (char*)d_ws;
  size_t off = 0;
  auto alloc = [&](size_t bytes) -> void* {
    void* ptr = (void*)(ws + off);
    off += (bytes + 255) & ~(size_t)255;
    return ptr;
  };
  __bf16* xb    = (__bf16*)alloc((size_t)N * 1024 * 2);   // x[:, :1024] bf16
  __bf16* p     = (__bf16*)alloc((size_t)N * 1024 * 2);   // (N,4,256) bf16
  __bf16* obar  = (__bf16*)alloc((size_t)N * 256 * 2);
  __bf16* fused = (__bf16*)alloc((size_t)N * 256 * 2);
  __bf16* h1b   = (__bf16*)alloc((size_t)N * 256 * 2);
  float*  hw    = (float*) alloc((size_t)N * 256 * 4);
  float*  h1    = (float*) alloc((size_t)N * 256 * 4);
  float*  deg   = (float*) alloc((size_t)N * 4);
  float*  dinv  = (float*) alloc((size_t)N * 4);
  __bf16* wgoT  = (__bf16*)alloc(512 * 256 * 2);          // (256,512)
  __bf16* wreT  = (__bf16*)alloc(256 * 256 * 2);          // (256,256)
  __bf16* wmeT  = (__bf16*)alloc(256 * 256 * 2);
  __bf16* wipB  = (__bf16*)alloc(768 * 256 * 2);          // in_proj  (768,256)
  __bf16* wopB  = (__bf16*)alloc(256 * 256 * 2);          // out_proj (256,256)
  __bf16* w1T   = (__bf16*)alloc(256 * 256 * 2);          // (256,256)
  __bf16* w2T   = (__bf16*)alloc(256 * 128 * 2);          // (128,256)

  // ---- conversions (weights transposed to N x K bf16; activations to bf16) ----
  conv_x<<<N, 256, 0, stream>>>(x, xb);
  conv_wt<<<(512 * 256 + 255) / 256, 256, 0, stream>>>(w_go, wgoT, 512, 256);
  conv_wt<<<(256 * 256 + 255) / 256, 256, 0, stream>>>(w_re, wreT, 256, 256);
  conv_wt<<<(256 * 256 + 255) / 256, 256, 0, stream>>>(w_me, wmeT, 256, 256);
  conv_w <<<(768 * 256 + 255) / 256, 256, 0, stream>>>(in_proj_w, wipB, 768 * 256);
  conv_w <<<(256 * 256 + 255) / 256, 256, 0, stream>>>(out_proj_w, wopB, 256 * 256);
  conv_wt<<<(256 * 256 + 255) / 256, 256, 0, stream>>>(w1, w1T, 256, 256);
  conv_wt<<<(256 * 128 + 255) / 256, 256, 0, stream>>>(w2, w2T, 256, 128);

  const int tilesM = N / 16;                 // 3125
  const dim3 blk(128);
  const dim3 grd256(8, (tilesM + 3) / 4);    // 256 output cols (2 tiles/wave)
  const dim3 grd128(4, (tilesM + 3) / 4);    // 128 output cols

  // p tokens 0..2 (relu projections), token 3 (gw outer product)
  gemm_bf16<true, true><<<grd256, blk, 0, stream>>>(xb,       1024, wgoT, 512, b_go, p,       1024, tilesM, 512);
  gemm_bf16<true, true><<<grd256, blk, 0, stream>>>(xb + 512, 1024, wreT, 256, b_re, p + 256, 1024, tilesM, 256);
  gemm_bf16<true, true><<<grd256, blk, 0, stream>>>(xb + 768, 1024, wmeT, 256, b_me, p + 512, 1024, tilesM, 256);
  gw_token<<<N, 256, 0, stream>>>(x, w_gw, b_gw, p);

  // fused QKV + attention + token-mean (q,k,v never touch HBM)
  qkv_attn<<<N / 4, 256, 0, stream>>>(p, wipB, in_proj_b, obar);

  // out_proj on the token-mean (mean commutes with linear layer)
  gemm_bf16<true, false><<<grd256, blk, 0, stream>>>(obar, 256, wopB, 256, out_proj_b, fused, 256, tilesM, 256);

  // degrees with self loops
  deg_init<<<(N + 255) / 256, 256, 0, stream>>>(deg, N);
  deg_count<<<(E + 255) / 256, 256, 0, stream>>>(ei, E, deg);
  dinv_k<<<(N + 255) / 256, 256, 0, stream>>>(deg, dinv, N);

  // GCN layer 1: h1 = relu(segsum(norm * (fused@w1)[src]) + b1)
  gemm_bf16<false, false><<<grd256, blk, 0, stream>>>(fused, 256, w1T, 256, nullptr, hw, 256, tilesM, 256);
  self_init<<<N, 256, 0, stream>>>(hw, dinv, h1, 256);
  edge_scatter<<<E, 256, 0, stream>>>(ei, E, hw, dinv, h1, 256);
  bias_act<true><<<N, 256, 0, stream>>>(h1, b1, h1b, 256);

  // GCN layer 2 -> d_out (N,128)
  gemm_bf16<false, false><<<grd128, blk, 0, stream>>>(h1b, 256, w2T, 256, nullptr, hw, 128, tilesM, 256);
  self_init<<<N, 128, 0, stream>>>(hw, dinv, out, 128);
  edge_scatter<<<E, 128, 0, stream>>>(ei, E, hw, dinv, out, 128);
  bias_act<false><<<N, 128, 0, stream>>>(out, b2, nullptr, 128);
}
